// SyntaxTreeLSTM_69260642615565
// MI455X (gfx1250) — matrix operand order repaired and checked by reading, not compiled, not run on hardware
//
#include <hip/hip_runtime.h>

#define BATCH  32768
#define TSTEPS 16

typedef __attribute__((ext_vector_type(16))) __bf16 v16bf;
typedef __attribute__((ext_vector_type(8)))  float  v8f;

union Frag { unsigned u[8]; uint4 q[2]; v16bf v; };

__device__ __forceinline__ unsigned short f2bf(float f) {
  union { float f; unsigned u; } v; v.f = f;
  unsigned r = v.u + 0x7FFFu + ((v.u >> 16) & 1u);
  return (unsigned short)(r >> 16);
}
__device__ __forceinline__ unsigned bfpack2(float lo, float hi) {
  union { float f; unsigned u; } a, b; a.f = lo; b.f = hi;
  unsigned ra = a.u + 0x7FFFu + ((a.u >> 16) & 1u);
  unsigned rb = b.u + 0x7FFFu + ((b.u >> 16) & 1u);
  return ((ra >> 16) & 0xFFFFu) | (rb & 0xFFFF0000u);
}
__device__ __forceinline__ float sigm(float x)  { return 1.f / (1.f + __expf(-x)); }
__device__ __forceinline__ float tanha(float x) { return 2.f * sigm(2.f * x) - 1.f; }
__device__ __forceinline__ float eluf(float x)  { return x > 0.f ? x : (__expf(x) - 1.f); }

__device__ __forceinline__ v8f wmma_bf16(v16bf a, v16bf b, v8f c) {
  return __builtin_amdgcn_wmma_f32_16x16x32_bf16(false, a, false, b, (short)0, c, false, false);
}

// ---- CDNA5 async global->LDS copy (16 B per lane), ASYNCcnt-tracked ----
__device__ __forceinline__ void async_cp16(unsigned ldsoff, const unsigned short* g) {
  asm volatile("global_load_async_to_lds_b128 %0, %1, off"
               :: "v"(ldsoff), "v"(g) : "memory");
}
__device__ __forceinline__ void wait_async0() {
  asm volatile("s_wait_asynccnt 0x0" ::: "memory");
}
__device__ __forceinline__ unsigned lds_off(const void* p) {
  return (unsigned)(size_t)p;  // low 32 bits of flat shared address = LDS offset
}

// B fragment (K=32 x N=16) from row-major [N][ldk] bf16 (torch [out][in] layout).
__device__ __forceinline__ v16bf load_bfrag(const unsigned short* W,
                                            int ldk, int ncol, int kb, int lane) {
  int n    = ncol + (lane & 15);
  int kadd = kb + ((lane & 16) ? 16 : 0);
  const uint4* p = (const uint4*)(W + n * ldk + kadd);
  Frag f; f.q[0] = p[0]; f.q[1] = p[1];
  return f.v;
}
// A fragment (M=16 x K=32) from row-major [rows][ldk] bf16.
__device__ __forceinline__ v16bf load_afrag_bf16(const unsigned short* X,
                                                 int ldk, int rbase, int kb, int lane) {
  int row = rbase + (lane & 15);
  int khi = (lane & 16) ? 16 : 0;
  const unsigned short* bp = X + row * ldk + kb + khi;
  Frag f;
  f.q[0] = *(const uint4*)(bp);
  f.q[1] = *(const uint4*)(bp + 8);
  return f.v;
}
// A fragment from fp32 source, converted to bf16 on the fly.
__device__ __forceinline__ v16bf load_afrag_f32(const float* __restrict__ X,
                                                int ldk, int rbase, int kb, int lane) {
  int row = rbase + (lane & 15);
  int khi = (lane & 16) ? 16 : 0;
  const float* bp = X + (size_t)row * ldk + kb + khi;
  float4 q0 = *(const float4*)(bp);
  float4 q1 = *(const float4*)(bp + 4);
  float4 q2 = *(const float4*)(bp + 8);
  float4 q3 = *(const float4*)(bp + 12);
  Frag f;
  f.u[0] = bfpack2(q0.x, q0.y); f.u[1] = bfpack2(q0.z, q0.w);
  f.u[2] = bfpack2(q1.x, q1.y); f.u[3] = bfpack2(q1.z, q1.w);
  f.u[4] = bfpack2(q2.x, q2.y); f.u[5] = bfpack2(q2.z, q2.w);
  f.u[6] = bfpack2(q3.x, q3.y); f.u[7] = bfpack2(q3.z, q3.w);
  return f.v;
}

// ---- cooperative async weight-slice staging (64 rows of a gate-column slice) ----
// R0 rows: 33 uint4 stride (264 halves), R1 rows: 17 uint4 stride (136 halves);
// row stride % 64 banks == 4 dwords -> conflict-free strided fragment reads.
// Base pointers are laundered through empty asm so the 24 per-slice address
// chains cannot be hoisted out of the step loop (round-4 showed hoist->spill).
__device__ __forceinline__ void stage_l1_async(const unsigned short* wa,
                                               const unsigned short* wb,
                                               int nb, int t, uint4* R0, uint4* R1) {
  asm volatile("" : "+v"(wa), "+v"(wb), "+v"(R0), "+v"(R1));
#pragma unroll
  for (int j = 0; j < 16; ++j) {  // Wih1 slice: 64 rows x 32 uint4
    int i = t + 128 * j, lr = i >> 5, c = i & 31, g = lr >> 4, r = lr & 15;
    async_cp16(lds_off(&R0[lr * 33 + c]), wa + (g * 128 + nb + r) * 256 + c * 8);
  }
#pragma unroll
  for (int j = 0; j < 8; ++j) {   // Whh1 slice: 64 rows x 16 uint4
    int i = t + 128 * j, lr = i >> 4, c = i & 15, g = lr >> 4, r = lr & 15;
    async_cp16(lds_off(&R1[lr * 17 + c]), wb + (g * 128 + nb + r) * 128 + c * 8);
  }
}
__device__ __forceinline__ void stage_l2_async(const unsigned short* wa,
                                               const unsigned short* wb,
                                               int nb, int t, uint4* R0, uint4* R1) {
  asm volatile("" : "+v"(wa), "+v"(wb), "+v"(R0), "+v"(R1));
#pragma unroll
  for (int j = 0; j < 8; ++j) {
    int i = t + 128 * j, lr = i >> 4, c = i & 15, g = lr >> 4, r = lr & 15;
    async_cp16(lds_off(&R0[lr * 33 + c]), wa + (g * 128 + nb + r) * 128 + c * 8);
    async_cp16(lds_off(&R1[lr * 17 + c]), wb + (g * 128 + nb + r) * 128 + c * 8);
  }
}

// ---------------- weight prep: fp32 -> bf16, fuse biases ----------------
__global__ __launch_bounds__(256) void prep_kernel(
    const float* __restrict__ wih1, const float* __restrict__ whh1,
    const float* __restrict__ wih2, const float* __restrict__ whh2,
    const float* __restrict__ hw,
    const float* __restrict__ bih1, const float* __restrict__ bhh1,
    const float* __restrict__ bih2, const float* __restrict__ bhh2,
    unsigned short* __restrict__ w1bf, unsigned short* __restrict__ whh1bf,
    unsigned short* __restrict__ wih2bf, unsigned short* __restrict__ whh2bf,
    unsigned short* __restrict__ headbf, float* __restrict__ bias1, float* __restrict__ bias2) {
  int i = blockIdx.x * 256 + threadIdx.x;
  if (i < 131072) { w1bf[i] = f2bf(wih1[i]); return; }
  i -= 131072;
  if (i < 65536) { whh1bf[i] = f2bf(whh1[i]); return; }
  i -= 65536;
  if (i < 65536) { wih2bf[i] = f2bf(wih2[i]); return; }
  i -= 65536;
  if (i < 65536) { whh2bf[i] = f2bf(whh2[i]); return; }
  i -= 65536;
  if (i < 1152) { headbf[i] = f2bf(hw[i]); return; }
  i -= 1152;
  if (i < 512) { bias1[i] = bih1[i] + bhh1[i]; bias2[i] = bih2[i] + bhh2[i]; }
}

// ---------------- conv stack -> bf16 [B,256] (upper 128 cols zero) ----------------
#define NB 8
__global__ __launch_bounds__(256) void conv_kernel(
    const float* __restrict__ x,
    const float* __restrict__ w1, const float* __restrict__ b1,
    const float* __restrict__ w2, const float* __restrict__ b2,
    const float* __restrict__ w3, const float* __restrict__ b3,
    unsigned short* __restrict__ xext) {
  __shared__ float xs[NB][20];
  __shared__ float a1[NB][32][14];
  __shared__ float a2[NB][64][8];
  __shared__ float w1s[32][8];
  __shared__ float b1s[32], b2s[64];
  const int t  = threadIdx.x;
  const int b0 = blockIdx.x * NB;
  for (int i = t; i < NB * 19; i += 256) { int e = i / 19, p = i % 19; xs[e][p] = x[(b0 + e) * 19 + p]; }
  for (int i = t; i < 32 * 7; i += 256) w1s[i / 7][i % 7] = w1[i];
  if (t < 32) b1s[t] = b1[t];
  if (t < 64) b2s[t] = b2[t];
  __syncthreads();
  for (int i = t; i < NB * 32 * 13; i += 256) {
    int e = i / 416, r = i % 416, co = r / 13, p = r % 13;
    float s = b1s[co];
#pragma unroll
    for (int k = 0; k < 7; ++k) s += xs[e][p + k] * w1s[co][k];
    a1[e][co][p] = eluf(s);
  }
  __syncthreads();
  for (int i = t; i < NB * 64 * 7; i += 256) {
    int e = i / 448, r = i % 448, co = r / 7, p = r % 7;
    float s = b2s[co];
    const float* wr = w2 + co * 224;
    for (int ci = 0; ci < 32; ++ci) {
#pragma unroll
      for (int k = 0; k < 7; ++k) s += a1[e][ci][p + k] * wr[ci * 7 + k];
    }
    a2[e][co][p] = eluf(s);
  }
  __syncthreads();
  for (int i = t; i < NB * 128; i += 256) {
    int e = i / 128, co = i % 128;
    float s = b3[co];
    const float* wr = w3 + co * 448;
    for (int ci = 0; ci < 64; ++ci) {
#pragma unroll
      for (int k = 0; k < 7; ++k) s += a2[e][ci][k] * wr[ci * 7 + k];
    }
    int b = b0 + e;
    xext[b * 256 + co]       = f2bf(eluf(s));
    xext[b * 256 + 128 + co] = 0;
  }
}

// ---------------- 2-layer LSTM, 17 steps, per-wave 16-row batch tile,
// block-shared double-buffered LDS weight slices via async global->LDS ----------------
__global__ __launch_bounds__(128, 1) void lstm_kernel(
    const float* __restrict__ ps, const unsigned short* __restrict__ xext,
    const unsigned short* __restrict__ w1bf, const unsigned short* __restrict__ whh1bf,
    const unsigned short* __restrict__ wih2bf, const unsigned short* __restrict__ whh2bf,
    const unsigned short* __restrict__ headbf, const float* __restrict__ bias1g,
    const float* __restrict__ bias2g, const float* __restrict__ head_b,
    float* __restrict__ out) {
  __shared__ __align__(16) uint4 wR0[2][64 * 33];              // 2 x 33 KB ih-slice
  __shared__ __align__(16) uint4 wR1[2][64 * 17];              // 2 x 17 KB hh-slice
  __shared__ __align__(16) unsigned short stage[4][16 * 136];  // per-wave D->A relayout
  __shared__ float biasL[2][512];
  const int t = threadIdx.x, wv = t >> 5, lane = t & 31;
  const int col = lane & 15, hi8 = (lane & 16) ? 8 : 0;
  for (int i = t; i < 512; i += 128) { biasL[0][i] = bias1g[i]; biasL[1][i] = bias2g[i]; }
  stage_l1_async(w1bf, whh1bf, 0, t, wR0[0], wR1[0]);  // pre-stage layer-1 slice 0
  wait_async0();
  __syncthreads();
  unsigned short* stg = stage[wv];
  const int rb = (blockIdx.x * 4 + wv) * 16;

  v16bf headB[4];
  {
    const int kadd0 = (lane & 16) ? 16 : 0;
#pragma unroll
    for (int k = 0; k < 4; ++k) {
      Frag f;
      if (col < 9) {
        const uint4* p = (const uint4*)(headbf + col * 128 + k * 32 + kadd0);
        f.q[0] = p[0]; f.q[1] = p[1];
      } else {
        f.q[0] = make_uint4(0, 0, 0, 0); f.q[1] = make_uint4(0, 0, 0, 0);
      }
      headB[k] = f.v;
    }
  }
  const float hb = (col < 9) ? head_b[col] : 0.f;

  v8f  c1[8], c2[8];
  v16bf h1A[4], h2A[4];
  const v8f zf = {0.f, 0.f, 0.f, 0.f, 0.f, 0.f, 0.f, 0.f};
  {
    Frag z; z.q[0] = make_uint4(0, 0, 0, 0); z.q[1] = make_uint4(0, 0, 0, 0);
#pragma unroll
    for (int k = 0; k < 4; ++k) { h1A[k] = z.v; h2A[k] = z.v; }
#pragma unroll
    for (int hc = 0; hc < 8; ++hc) { c1[hc] = zf; c2[hc] = zf; }
  }

#pragma unroll 1
  for (int s = 0; s < 17; ++s) {
    asm volatile("" ::: "memory");  // keep step-invariant loads inside the loop

    v16bf xA[8];
    if (s == 0) {
#pragma unroll
      for (int k = 0; k < 8; ++k) xA[k] = load_afrag_bf16(xext, 256, rb, k * 32, lane);
    } else {
      const float* xp = ps + (size_t)(s - 1) * BATCH * 256;
#pragma unroll
      for (int k = 0; k < 8; ++k) xA[k] = load_afrag_f32(xp, 256, rb, k * 32, lane);
    }

    // ---- layer 1 ----
#pragma unroll
    for (int hc = 0; hc < 8; ++hc) {
      // Async-stage the next slice into the other buffer; safe: the barrier that
      // ended iteration hc-1 retired all reads of that buffer.
      if (hc < 7) stage_l1_async(w1bf, whh1bf, (hc + 1) * 16, t, wR0[(hc + 1) & 1], wR1[(hc + 1) & 1]);
      else        stage_l2_async(wih2bf, whh2bf, 0, t, wR0[0], wR1[0]);
      const unsigned short* R0c = (const unsigned short*)wR0[hc & 1];
      const unsigned short* R1c = (const unsigned short*)wR1[hc & 1];
      v8f acc[4];
      v16bf buf[2][4];
#pragma unroll
      for (int g = 0; g < 4; ++g) buf[0][g] = load_bfrag(R0c, 264, g * 16, 0, lane);
#pragma unroll
      for (int k = 0; k < 12; ++k) {
        const int cur = k & 1, nxt = cur ^ 1;
        if (k < 11) {
          const int kn = k + 1;
#pragma unroll
          for (int g = 0; g < 4; ++g)
            buf[nxt][g] = (kn < 8)
                ? load_bfrag(R0c, 264, g * 16, kn * 32, lane)
                : load_bfrag(R1c, 136, g * 16, (kn - 8) * 32, lane);
        }
        v16bf a = (k < 8) ? xA[k] : h1A[k - 8];
#pragma unroll
        for (int g = 0; g < 4; ++g) acc[g] = wmma_bf16(a, buf[cur][g], (k == 0) ? zf : acc[g]);
      }
#pragma unroll
      for (int e = 0; e < 8; ++e) {
        float iv = sigm(acc[0][e] + biasL[0][0 * 128 + hc * 16 + col]);
        float fv = sigm(acc[1][e] + biasL[0][1 * 128 + hc * 16 + col]);
        float gv = tanha(acc[2][e] + biasL[0][2 * 128 + hc * 16 + col]);
        float ov = sigm(acc[3][e] + biasL[0][3 * 128 + hc * 16 + col]);
        float cv = fv * c1[hc][e] + iv * gv;
        c1[hc][e] = cv;
        stg[(hi8 + e) * 136 + hc * 16 + col] = f2bf(ov * tanha(cv));
      }
      wait_async0();
      __syncthreads();
    }
#pragma unroll
    for (int k = 0; k < 4; ++k) h1A[k] = load_afrag_bf16(stg, 136, 0, k * 32, lane);

    // ---- layer 2 ----
#pragma unroll
    for (int hc = 0; hc < 8; ++hc) {
      if (hc < 7) stage_l2_async(wih2bf, whh2bf, (hc + 1) * 16, t, wR0[(hc + 1) & 1], wR1[(hc + 1) & 1]);
      else        stage_l1_async(w1bf, whh1bf, 0, t, wR0[0], wR1[0]);  // next step's slice 0
      const unsigned short* R0c = (const unsigned short*)wR0[hc & 1];
      const unsigned short* R1c = (const unsigned short*)wR1[hc & 1];
      v8f acc[4];
      v16bf buf[2][4];
#pragma unroll
      for (int g = 0; g < 4; ++g) buf[0][g] = load_bfrag(R0c, 264, g * 16, 0, lane);
#pragma unroll
      for (int k = 0; k < 8; ++k) {
        const int cur = k & 1, nxt = cur ^ 1;
        if (k < 7) {
          const int kn = k + 1;
#pragma unroll
          for (int g = 0; g < 4; ++g)
            buf[nxt][g] = (kn < 4)
                ? load_bfrag(R0c, 264, g * 16, kn * 32, lane)
                : load_bfrag(R1c, 136, g * 16, (kn - 4) * 32, lane);
        }
        v16bf a = (k < 4) ? h1A[k] : h2A[k - 4];
#pragma unroll
        for (int g = 0; g < 4; ++g) acc[g] = wmma_bf16(a, buf[cur][g], (k == 0) ? zf : acc[g]);
      }
#pragma unroll
      for (int e = 0; e < 8; ++e) {
        float iv = sigm(acc[0][e] + biasL[1][0 * 128 + hc * 16 + col]);
        float fv = sigm(acc[1][e] + biasL[1][1 * 128 + hc * 16 + col]);
        float gv = tanha(acc[2][e] + biasL[1][2 * 128 + hc * 16 + col]);
        float ov = sigm(acc[3][e] + biasL[1][3 * 128 + hc * 16 + col]);
        float cv = fv * c2[hc][e] + iv * gv;
        c2[hc][e] = cv;
        stg[(hi8 + e) * 136 + hc * 16 + col] = f2bf(ov * tanha(cv));
      }
      wait_async0();
      __syncthreads();
    }
#pragma unroll
    for (int k = 0; k < 4; ++k) h2A[k] = load_afrag_bf16(stg, 136, 0, k * 32, lane);

    // ---- head ----
    v8f ha = zf;
#pragma unroll
    for (int k = 0; k < 4; ++k) ha = wmma_bf16(h2A[k], headB[k], ha);
    if (col < 9) {
      float* op = out + ((size_t)s * BATCH + (size_t)(rb + hi8)) * 9 + col;
      const float hbv = hb;
#pragma unroll
      for (int r = 0; r < 8; ++r) op[(size_t)r * 9] = ha[r] + hbv;
    }
  }
}

extern "C" void kernel_launch(void* const* d_in, const int* in_sizes, int n_in,
                              void* d_out, int out_size, void* d_ws, size_t ws_size,
                              hipStream_t stream) {
  (void)in_sizes; (void)n_in; (void)out_size; (void)ws_size;
  const float* x    = (const float*)d_in[0];
  const float* ps   = (const float*)d_in[1];
  const float* c1w  = (const float*)d_in[2];
  const float* c1b  = (const float*)d_in[3];
  const float* c2w  = (const float*)d_in[4];
  const float* c2b  = (const float*)d_in[5];
  const float* c3w  = (const float*)d_in[6];
  const float* c3b  = (const float*)d_in[7];
  const float* wih1 = (const float*)d_in[8];
  const float* whh1 = (const float*)d_in[9];
  const float* bih1 = (const float*)d_in[10];
  const float* bhh1 = (const float*)d_in[11];
  const float* wih2 = (const float*)d_in[12];
  const float* whh2 = (const float*)d_in[13];
  const float* bih2 = (const float*)d_in[14];
  const float* bhh2 = (const float*)d_in[15];
  const float* hw   = (const float*)d_in[16];
  const float* hbv  = (const float*)d_in[17];
  float* out = (float*)d_out;

  char* ws = (char*)d_ws;
  unsigned short* w1bf   = (unsigned short*)(ws);
  unsigned short* whh1bf = (unsigned short*)(ws + 262144);
  unsigned short* wih2bf = (unsigned short*)(ws + 393216);
  unsigned short* whh2bf = (unsigned short*)(ws + 524288);
  unsigned short* headbf = (unsigned short*)(ws + 655360);
  float*          bias1  = (float*)(ws + 659456);
  float*          bias2  = (float*)(ws + 661504);
  unsigned short* xext   = (unsigned short*)(ws + 663552);

  prep_kernel<<<(329344 + 255) / 256, 256, 0, stream>>>(
      wih1, whh1, wih2, whh2, hw, bih1, bhh1, bih2, bhh2,
      w1bf, whh1bf, wih2bf, whh2bf, headbf, bias1, bias2);
  conv_kernel<<<BATCH / NB, 256, 0, stream>>>(x, c1w, c1b, c2w, c2b, c3w, c3b, xext);
  lstm_kernel<<<BATCH / 64, 128, 0, stream>>>(
      ps, xext, w1bf, whh1bf, wih2bf, whh2bf, headbf, bias1, bias2, hbv, out);
}
